// PCTokenizer_91336774516775
// MI455X (gfx1250) — matrix operand order repaired and checked by reading, not compiled, not run on hardware
//
#include <hip/hip_runtime.h>
#include <hip/hip_bf16.h>
#include <math.h>

// ---------------- problem constants ----------------
#define BB      32
#define NN      8192
#define GG      128
#define KK      32
#define ENC_C   384
#define NUM_VIS 52          // G - int(0.6*G) = 128 - 76
#define EPSBN   1e-5f

// output region offsets (floats)
#define OUT_TOKVIS 0
#define OUT_POS    (BB*NUM_VIS*ENC_C)                 // 638976
#define OUT_MASK   (OUT_POS + BB*NUM_VIS*ENC_C)       // 1277952
#define OUT_CENTER (OUT_MASK + BB*GG)                 // 1282048
#define OUT_NEIGH  (OUT_CENTER + BB*GG*3)             // 1294336

// workspace layout (bytes)
#define WS_TOKENS  0                                   // B*G*384 f32 = 6291456 B
#define WS_W2H     (BB*GG*ENC_C*4)                     // 6291456
#define WS_W3H     (WS_W2H + 256*128*2)                // +65536
#define WS_W4H     (WS_W3H + 512*512*2)                // +524288

typedef __attribute__((ext_vector_type(16))) _Float16 v16h;
typedef __attribute__((ext_vector_type(8)))  _Float16 v8h;
typedef __attribute__((ext_vector_type(8)))  float    v8f;
typedef __attribute__((ext_vector_type(4)))  unsigned int u32x4;
typedef __attribute__((ext_vector_type(8)))  int      i32x8;
typedef __attribute__((ext_vector_type(4)))  int      i32x4;

// address-space qualified int4 for the async-copy builtin
typedef __attribute__((address_space(1))) i32x4 as1_i32x4;
typedef __attribute__((address_space(3))) i32x4 as3_i32x4;

// ---------------- CDNA5 async / TDM helpers ----------------
// per-lane 16B global -> LDS async copy (ASYNCcnt)
__device__ __forceinline__ void async_b128(const void* g, void* l) {
    __builtin_amdgcn_global_load_async_to_lds_b128(
        (as1_i32x4*)(g), (as3_i32x4*)(l), 0, 0);
}
__device__ __forceinline__ void wait_async0() {
    asm volatile("s_wait_asynccnt 0x0" ::: "memory");
}

// 1-D contiguous TDM copy global -> LDS, `bytes` must be a multiple of 4.
// D#: data_size=4B, tile_dim0 = bytes/4 (<=65535), tensor == tile, no padding.
__device__ __forceinline__ void tdm_load_1d(const void* gsrc, void* ldst,
                                            unsigned bytes) {
    unsigned n4 = bytes >> 2;
    unsigned lds_off = (unsigned)(unsigned long long)
        (__attribute__((address_space(3))) char*)(ldst);
    unsigned long long ga = (unsigned long long)(gsrc);
    u32x4 g0;
    g0[0] = 1u;                                   // count=1, user descriptor
    g0[1] = lds_off;                              // lds_addr (bytes)
    g0[2] = (unsigned)ga;                         // global_addr[31:0]
    g0[3] = (unsigned)(ga >> 32) | (2u << 30);    // global_addr[56:32] | type=2
    i32x8 g1;
    g1[0] = (int)(2u << 16);                      // wg_mask=0, data_size=4B
    g1[1] = (int)((n4 & 0xffffu) << 16);          // tensor_dim0[15:0]
    g1[2] = (int)(((n4 >> 16) & 0xffffu) | (1u << 16)); // dim0 hi | tensor_dim1=1
    g1[3] = (int)((n4 & 0xffffu) << 16);          // tensor_dim1 hi=0 | tile_dim0
    g1[4] = 1;                                    // tile_dim1=1, tile_dim2=0
    g1[5] = (int)n4;                              // tensor_dim0_stride[31:0]
    g1[6] = 0;                                    // stride hi | dim1_stride lo
    g1[7] = 0;
    i32x4 z = {0, 0, 0, 0};
#if __clang_major__ >= 23
    i32x8 z8 = {0, 0, 0, 0, 0, 0, 0, 0};
    __builtin_amdgcn_tensor_load_to_lds(g0, g1, z, z, z8, 0);
#else
    __builtin_amdgcn_tensor_load_to_lds(g0, g1, z, z, 0);
#endif
}

// ---------------- WMMA fragment helpers (CDNA5 16-bit layouts) ----------------
__device__ __forceinline__ v16h frag_a(const _Float16* base, int ld, int lane) {
    int m  = lane & 15;
    int kb = (lane >> 4) << 3;
    const _Float16* p = base + m * ld + kb;
    v8h lo = *(const v8h*)(p);
    v8h hi = *(const v8h*)(p + 16);
    v16h a;
#pragma unroll
    for (int e = 0; e < 8; ++e) { a[e] = lo[e]; a[e + 8] = hi[e]; }
    return a;
}
__device__ __forceinline__ v16h frag_b(const _Float16* base, int ld, int lane) {
    int n  = lane & 15;
    int kb = (lane >> 4) << 3;
    const _Float16* p = base + n * ld + kb;
    v8h lo = *(const v8h*)(p);
    v8h hi = *(const v8h*)(p + 16);
    v16h b;
#pragma unroll
    for (int e = 0; e < 8; ++e) { b[e] = lo[e]; b[e + 8] = hi[e]; }
    return b;
}
__device__ __forceinline__ v8f wmma_acc(v16h a, v16h b, v8f c) {
    return __builtin_amdgcn_wmma_f32_16x16x32_f16(false, a, false, b, (short)0, c,
                                                  false, false);
}

// ---------------- kernel 0: one-time f32 -> f16 weight conversion ----------------
__global__ __launch_bounds__(256) void convert_weights_kernel(
    const float* __restrict__ w2, const float* __restrict__ w3,
    const float* __restrict__ w4, _Float16* __restrict__ w2h,
    _Float16* __restrict__ w3h, _Float16* __restrict__ w4h) {
    const int i = blockIdx.x * 256 + threadIdx.x;
    const int stride = gridDim.x * 256;
    for (int j = i; j < 256 * 128; j += stride) w2h[j] = (_Float16)w2[j];
    for (int j = i; j < 512 * 512; j += stride) w3h[j] = (_Float16)w3[j];
    for (int j = i; j < 384 * 512; j += stride) w4h[j] = (_Float16)w4[j];
}

// ---------------- kernel 1: farthest point sampling ----------------
__global__ __launch_bounds__(256) void fps_kernel(const float* __restrict__ xyz,
                                                  float* __restrict__ center_out) {
    extern __shared__ char smem[];
    float* sx   = (float*)smem;            // [NN*3]
    float* dist = sx + NN * 3;             // [NN]
    float* redV = dist + NN;               // [256]
    int*   redI = (int*)(redV + 256);      // [256]

    const int b = blockIdx.x, t = threadIdx.x;
    const float* xb = xyz + (long)b * NN * 3;
    // async-stage the whole batch point cloud (96 KB) into LDS
    for (int i = t; i < (NN * 3 * 4) / 16; i += 256)
        async_b128((const char*)xb + i * 16, (char*)sx + i * 16);
    for (int i = t; i < NN; i += 256) dist[i] = __builtin_inff();
    wait_async0();
    __syncthreads();

    int far = 0;
    for (int g = 0; g < GG; ++g) {
        float cx = sx[far * 3], cy = sx[far * 3 + 1], cz = sx[far * 3 + 2];
        if (t == 0) {
            long o = ((long)b * GG + g) * 3;
            center_out[o] = cx; center_out[o + 1] = cy; center_out[o + 2] = cz;
        }
        float bv = -1.0f; int bi = 0;
        for (int i = t; i < NN; i += 256) {
            float dx = sx[i * 3] - cx, dy = sx[i * 3 + 1] - cy, dz = sx[i * 3 + 2] - cz;
            float nd = fminf(dist[i], dx * dx + dy * dy + dz * dz);
            dist[i] = nd;
            if (nd > bv) { bv = nd; bi = i; }
        }
        redV[t] = bv; redI[t] = bi;
        __syncthreads();
        for (int s = 128; s > 0; s >>= 1) {
            if (t < s) {
                if (redV[t + s] > redV[t] ||
                    (redV[t + s] == redV[t] && redI[t + s] < redI[t])) {
                    redV[t] = redV[t + s]; redI[t] = redI[t + s];
                }
            }
            __syncthreads();
        }
        far = redI[0];
        __syncthreads();
    }
}

// ---------------- kernel 2: KNN (32 smallest d2, ascending, stable) ----------------
__global__ __launch_bounds__(256) void knn_kernel(const float* __restrict__ xyz,
                                                  const float* __restrict__ center,
                                                  float* __restrict__ neigh_out) {
    __shared__ float d2[NN];
    __shared__ float redV[256];
    __shared__ int   redI[256];
    __shared__ int   sel[KK];

    const int bg = blockIdx.x, t = threadIdx.x;
    const int b = bg >> 7;  // / GG
    const float* xb = xyz + (long)b * NN * 3;
    const float cx = center[bg * 3], cy = center[bg * 3 + 1], cz = center[bg * 3 + 2];

    for (int i = t; i < NN; i += 256) {
        float dx = xb[i * 3] - cx, dy = xb[i * 3 + 1] - cy, dz = xb[i * 3 + 2] - cz;
        d2[i] = dx * dx + dy * dy + dz * dz;
    }
    __syncthreads();

    for (int k = 0; k < KK; ++k) {
        float bv = __builtin_inff(); int bi = 0x7fffffff;
        for (int i = t; i < NN; i += 256) {
            float v = d2[i];
            if (v < bv || (v == bv && i < bi)) { bv = v; bi = i; }
        }
        redV[t] = bv; redI[t] = bi;
        __syncthreads();
        for (int s = 128; s > 0; s >>= 1) {
            if (t < s) {
                if (redV[t + s] < redV[t] ||
                    (redV[t + s] == redV[t] && redI[t + s] < redI[t])) {
                    redV[t] = redV[t + s]; redI[t] = redI[t + s];
                }
            }
            __syncthreads();
        }
        if (t == 0) { sel[k] = redI[0]; d2[redI[0]] = __builtin_inff(); }
        __syncthreads();
    }
    if (t < KK) {
        int j = sel[t];
        long o = ((long)bg * KK + t) * 3;
        neigh_out[o]     = xb[j * 3]     - cx;
        neigh_out[o + 1] = xb[j * 3 + 1] - cy;
        neigh_out[o + 2] = xb[j * 3 + 2] - cz;
    }
}

// ---------------- kernel 3: per-group PointNet via WMMA + TDM panels ----------------
// One block = 2 groups (M = 64 rows). 8 waves = 4 M-tiles x 2 N-subtiles.
// Wave 0 drives the Tensor Data Mover to double-buffer 32 KB f16 weight panels.
__global__ __launch_bounds__(256) void mlp_kernel(
    const float* __restrict__ neigh,
    const float* __restrict__ w1, const float* __restrict__ b1,
    const float* __restrict__ g1, const float* __restrict__ be1,
    const float* __restrict__ m1, const float* __restrict__ v1,
    const _Float16* __restrict__ w2h, const float* __restrict__ b2,
    const _Float16* __restrict__ w3h, const float* __restrict__ b3,
    const float* __restrict__ g2, const float* __restrict__ be2,
    const float* __restrict__ m2, const float* __restrict__ v2,
    const _Float16* __restrict__ w4h, const float* __restrict__ b4,
    float* __restrict__ tokens) {
    extern __shared__ char smem[];
    _Float16* CAT  = (_Float16*)smem;          // [64][512] concat(fg, f2)
    _Float16* F3   = CAT + 64 * 512;           // [64][512]
    _Float16* F1   = F3 + 64 * 512;            // [64][128]
    _Float16* W2S  = F1 + 64 * 128;            // [256][128]
    _Float16* PAN0 = W2S + 256 * 128;          // [32][512]
    _Float16* PAN1 = PAN0 + 32 * 512;          // [32][512]
    float*    SCR  = (float*)(PAN1 + 32 * 512);// [64][32]
    float*    PG   = SCR + 64 * 32;            // [64][3]
    float*    S1   = PG + 64 * 3;              // [128]
    float*    T1   = S1 + 128;                 // [128]

    const int t = threadIdx.x;
    const int lane = t & 31;
    const int w = t >> 5;
    const int bb = blockIdx.x;
    const int b = bb >> 6;
    const int g0 = (bb & 63) << 1;
    const long grp = (long)b * GG + g0;

    // kick off TDM: w2 (64 KB) then first w3 panel (32 KB); overlap with stage 1
    if (w == 0) {
        tdm_load_1d(w2h, W2S, 256 * 128 * 2);
        tdm_load_1d(w3h, PAN0, 32 * 512 * 2);
    }

    // stage inputs
    const float* nb = neigh + grp * KK * 3;
    for (int i = t; i < 64 * 3; i += 256) PG[i] = nb[i];
    if (t < 128) {
        float s = g1[t] * rsqrtf(v1[t] + EPSBN);
        S1[t] = s; T1[t] = be1[t] - m1[t] * s;
    }
    __syncthreads();

    // stage 1: f1 = relu(bn1(pg @ w1^T + b1))   (K=3, VALU)
    for (int i = t; i < 64 * 128; i += 256) {
        int r = i >> 7, c = i & 127;
        float x = PG[r * 3] * w1[c * 3] + PG[r * 3 + 1] * w1[c * 3 + 1] +
                  PG[r * 3 + 2] * w1[c * 3 + 2] + b1[c];
        x = x * S1[c] + T1[c];
        F1[i] = (_Float16)fmaxf(x, 0.0f);
    }
    if (w == 0) __builtin_amdgcn_s_wait_tensorcnt((short)1);  // W2S landed
    __syncthreads();

    // stage 2: f2 = f1 @ w2^T + b2  ->  CAT right half (cols 256..511)
    {
        const int mt2 = w & 3;
        for (int nt = (w >> 2); nt < 16; nt += 2) {
            v8f acc = {};
#pragma unroll
            for (int k0 = 0; k0 < 128; k0 += 32) {
                v16h a  = frag_a(F1 + mt2 * 16 * 128 + k0, 128, lane);
                v16h bm = frag_b(W2S + nt * 16 * 128 + k0, 128, lane);
                acc = wmma_acc(a, bm, acc);
            }
            const int n = nt * 16 + (lane & 15);
            const float bias = b2[n];
            const int mrb = mt2 * 16 + ((lane >> 4) << 3);
#pragma unroll
            for (int r = 0; r < 8; ++r)
                CAT[(mrb + r) * 512 + 256 + n] = (_Float16)(acc[r] + bias);
        }
    }
    __syncthreads();

    // group max-pool -> broadcast into CAT left half (cols 0..255)
    for (int i = t; i < 2 * 256; i += 256) {
        int g = i >> 8, c = i & 255, rb = g * 32;
        float mx = -3.4e38f;
        for (int r = 0; r < 32; ++r)
            mx = fmaxf(mx, (float)CAT[(rb + r) * 512 + 256 + c]);
        _Float16 mh = (_Float16)mx;
        for (int r = 0; r < 32; ++r) CAT[(rb + r) * 512 + c] = mh;
    }

    const int mt = w & 3, ns = w >> 2;

    // stages 3+4 fused: 28 double-buffered 32-col panels
    //   p = 0..15 : f3 = relu(bn2(CAT @ w3^T + b3))          -> F3
    //   p = 16..27: f4 = F3 @ w4^T + b4, group-max -> tokens
    for (int p = 0; p < 28; ++p) {
        const _Float16* cur = (p & 1) ? PAN1 : PAN0;
        _Float16*       nxt = (p & 1) ? PAN0 : PAN1;
        if (w == 0) __builtin_amdgcn_s_wait_tensorcnt((short)0);  // panel p ready
        __syncthreads();  // panel visible to all waves; prev panel reads retired
        if (w == 0 && p + 1 < 28) {
            const _Float16* src = (p + 1 < 16) ? (w3h + (long)(p + 1) * 32 * 512)
                                               : (w4h + (long)(p - 15) * 32 * 512);
            tdm_load_1d(src, nxt, 32 * 512 * 2);  // DMA next panel during compute
        }
        const _Float16* A = (p < 16) ? CAT : F3;
        v8f acc = {};
#pragma unroll 4
        for (int k0 = 0; k0 < 512; k0 += 32) {
            v16h a  = frag_a(A + mt * 16 * 512 + k0, 512, lane);
            v16h bm = frag_b(cur + ns * 16 * 512 + k0, 512, lane);
            acc = wmma_acc(a, bm, acc);
        }
        const int mrb = mt * 16 + ((lane >> 4) << 3);
        if (p < 16) {
            const int n = p * 32 + ns * 16 + (lane & 15);
            const float s = g2[n] * rsqrtf(v2[n] + EPSBN);
            const float sh = be2[n] - m2[n] * s;
            const float bias = b3[n];
#pragma unroll
            for (int r = 0; r < 8; ++r) {
                float x = (acc[r] + bias) * s + sh;
                F3[(mrb + r) * 512 + n] = (_Float16)fmaxf(x, 0.0f);
            }
        } else {
            const int q = p - 16;
            const int nl = ns * 16 + (lane & 15);
            const float bias = b4[q * 32 + nl];
#pragma unroll
            for (int r = 0; r < 8; ++r) SCR[(mrb + r) * 32 + nl] = acc[r] + bias;
            __syncthreads();
            if (t < 64) {
                int g = t >> 5, c = t & 31;
                float mx = -3.4e38f;
                for (int r = 0; r < 32; ++r)
                    mx = fmaxf(mx, SCR[(g * 32 + r) * 32 + c]);
                tokens[(grp + g) * ENC_C + q * 32 + c] = mx;
            }
        }
    }
}

// ---------------- kernel 4: masking gather + pos embed ----------------
__global__ __launch_bounds__(256) void finalize_kernel(
    const unsigned char* __restrict__ mask, const float* __restrict__ tokens,
    const float* __restrict__ center,
    const float* __restrict__ pw1, const float* __restrict__ pb1,
    const float* __restrict__ pw2, const float* __restrict__ pb2,
    float* __restrict__ tokvis, float* __restrict__ pos,
    float* __restrict__ maskf) {
    __shared__ int   vis[NUM_VIS];
    __shared__ float cen[NUM_VIS * 3];
    __shared__ float h[NUM_VIS * 128];

    const int b = blockIdx.x, t = threadIdx.x;
    const unsigned char* mb = mask + b * GG;
    if (t == 0) {
        int c = 0;
        for (int g = 0; g < GG && c < NUM_VIS; ++g)
            if (!mb[g]) vis[c++] = g;   // stable argsort of 0/1 keys: zeros first
    }
    if (t < GG) maskf[b * GG + t] = mb[t] ? 1.0f : 0.0f;
    __syncthreads();

    for (int i = t; i < NUM_VIS * 3; i += 256) {
        int v = i / 3, d = i % 3;
        cen[i] = center[((long)b * GG + vis[v]) * 3 + d];
    }
    for (int i = t; i < NUM_VIS * ENC_C; i += 256) {
        int v = i / ENC_C, c = i % ENC_C;
        tokvis[((long)b * NUM_VIS + v) * ENC_C + c] =
            tokens[((long)b * GG + vis[v]) * ENC_C + c];
    }
    __syncthreads();

    for (int i = t; i < NUM_VIS * 128; i += 256) {
        int v = i >> 7, c = i & 127;
        float x = cen[v * 3] * pw1[c * 3] + cen[v * 3 + 1] * pw1[c * 3 + 1] +
                  cen[v * 3 + 2] * pw1[c * 3 + 2] + pb1[c];
        h[i] = 0.5f * x * (1.0f + erff(x * 0.70710678118654752f));  // exact GELU
    }
    __syncthreads();

    for (int i = t; i < NUM_VIS * ENC_C; i += 256) {
        int v = i / ENC_C, c = i % ENC_C;
        float acc = pb2[c];
        const float* hv = h + v * 128;
        const float* wr = pw2 + c * 128;
#pragma unroll 4
        for (int k = 0; k < 128; ++k) acc += hv[k] * wr[k];
        pos[((long)b * NUM_VIS + v) * ENC_C + c] = acc;
    }
}

// ---------------- host launcher ----------------
extern "C" void kernel_launch(void* const* d_in, const int* in_sizes, int n_in,
                              void* d_out, int out_size, void* d_ws, size_t ws_size,
                              hipStream_t stream) {
    const float*         xyz  = (const float*)d_in[0];
    const unsigned char* mask = (const unsigned char*)d_in[1];  // numpy bool
    const float* w1  = (const float*)d_in[2];
    const float* b1  = (const float*)d_in[3];
    const float* g1  = (const float*)d_in[4];
    const float* be1 = (const float*)d_in[5];
    const float* m1  = (const float*)d_in[6];
    const float* v1  = (const float*)d_in[7];
    const float* w2  = (const float*)d_in[8];
    const float* b2  = (const float*)d_in[9];
    const float* w3  = (const float*)d_in[10];
    const float* b3  = (const float*)d_in[11];
    const float* g2  = (const float*)d_in[12];
    const float* be2 = (const float*)d_in[13];
    const float* m2  = (const float*)d_in[14];
    const float* v2  = (const float*)d_in[15];
    const float* w4  = (const float*)d_in[16];
    const float* b4  = (const float*)d_in[17];
    const float* pw1 = (const float*)d_in[18];
    const float* pb1 = (const float*)d_in[19];
    const float* pw2 = (const float*)d_in[20];
    const float* pb2 = (const float*)d_in[21];

    float* out        = (float*)d_out;
    float* out_tokvis = out + OUT_TOKVIS;
    float* out_pos    = out + OUT_POS;
    float* out_mask   = out + OUT_MASK;
    float* out_center = out + OUT_CENTER;
    float* out_neigh  = out + OUT_NEIGH;

    char* ws = (char*)d_ws;
    float*    ws_tokens = (float*)(ws + WS_TOKENS);
    _Float16* w2h = (_Float16*)(ws + WS_W2H);
    _Float16* w3h = (_Float16*)(ws + WS_W3H);
    _Float16* w4h = (_Float16*)(ws + WS_W4H);

    convert_weights_kernel<<<128, 256, 0, stream>>>(w2, w3, w4, w2h, w3h, w4h);

    const size_t fps_lds = (size_t)(NN * 3 + NN + 256 + 256) * 4;  // ~133 KB
    fps_kernel<<<BB, 256, fps_lds, stream>>>(xyz, out_center);

    knn_kernel<<<BB * GG, 256, 0, stream>>>(xyz, out_center, out_neigh);

    const size_t mlp_lds =
        (size_t)(64 * 512 + 64 * 512 + 64 * 128 + 256 * 128 + 2 * 32 * 512) * 2 +
        (size_t)(64 * 32 + 64 * 3 + 128 + 128) * 4;                 // ~282 KB
    mlp_kernel<<<BB * GG / 2, 256, mlp_lds, stream>>>(
        out_neigh, w1, b1, g1, be1, m1, v1, w2h, b2, w3h, b3, g2, be2, m2, v2,
        w4h, b4, ws_tokens);

    finalize_kernel<<<BB, 256, 0, stream>>>(mask, ws_tokens, out_center, pw1, pb1,
                                            pw2, pb2, out_tokvis, out_pos, out_mask);
}
